// LSTM_22874995818703
// MI455X (gfx1250) — matrix operand order repaired and checked by reading, not compiled, not run on hardware
//
#include <hip/hip_runtime.h>
#include <stddef.h>

// ---------------------------------------------------------------------------
// LSTM cell fused forward for MI455X (gfx1250), bf16 WMMA + TDM staging.
//   gates GEMM:  G = [x|h] @ [Wx;Wh]^T  (M=16384, N=4096 interleaved, K=1536)
//   fused epilogue: sigmoid/tanh gates, c_new, h_new
//   out GEMM:    out = h_new @ Wout^T + bout (M=16384, N=512, K=1024)
// Tiles move global->LDS via tensor_load_to_lds (Tensor Data Mover), double
// buffered, tracked with TENSORcnt. Block tile 256(M)x128(N); each of the 8
// wave32 waves computes a 64x64 tile = 16 v_wmma_f32_16x16x32_bf16 per K-step.
// ---------------------------------------------------------------------------

typedef __bf16 bf16_t;
typedef __bf16 v16bf __attribute__((ext_vector_type(16)));
typedef __bf16 v8bf  __attribute__((ext_vector_type(8)));
typedef float  v8f   __attribute__((ext_vector_type(8)));
typedef float  v4f   __attribute__((ext_vector_type(4)));
typedef int    v4i   __attribute__((ext_vector_type(4)));
typedef int    v8i   __attribute__((ext_vector_type(8)));

#define B_  16384
#define I_  512
#define H_  1024
#define O_  512
#define KG  (I_ + H_)   // 1536, K of gates GEMM
#define NG  (4 * H_)    // 4096, N of gates GEMM (n = unit*4 + gate)

__device__ __forceinline__ bf16_t f2bf(float f) {
  unsigned u = __float_as_uint(f);
  unsigned r = (u + 0x7FFFu + ((u >> 16) & 1u)) >> 16;
  unsigned short s = (unsigned short)r;
  return __builtin_bit_cast(bf16_t, s);
}
__device__ __forceinline__ float fsig(float x)  { return 1.0f / (1.0f + __expf(-x)); }
__device__ __forceinline__ float ftanh_(float x){ float e = __expf(2.0f * x); return 1.0f - 2.0f / (e + 1.0f); }

__device__ __forceinline__ v8bf cvt8(const float* src) {
  v4f a = *(const v4f*)src;
  v4f b = *(const v4f*)(src + 4);
  v8bf r;
#pragma unroll
  for (int e = 0; e < 4; ++e) { r[e] = f2bf(a[e]); r[e + 4] = f2bf(b[e]); }
  return r;
}

// ------------------------------ pack kernels -------------------------------
// 8 elements per thread: 2x global_load_b128 in, 1x 16B store out.

__global__ __launch_bounds__(256) void pack_a_kernel(
    const float* __restrict__ inp, const float* __restrict__ h,
    bf16_t* __restrict__ Apack) {
  const size_t total8 = (size_t)B_ * KG / 8;
  const int kchunks = KG / 8;   // 192
  for (size_t idx = (size_t)blockIdx.x * 256 + threadIdx.x; idx < total8;
       idx += (size_t)gridDim.x * 256) {
    size_t b = idx / kchunks;
    int k = (int)(idx % kchunks) * 8;
    const float* src = (k < I_) ? (inp + b * I_ + k) : (h + b * H_ + (k - I_));
    *(v8bf*)(Apack + idx * 8) = cvt8(src);
  }
}

__global__ __launch_bounds__(256) void pack_w_kernel(
    const float* __restrict__ Wx, const float* __restrict__ Wh,
    bf16_t* __restrict__ Wg) {
  const size_t total8 = (size_t)NG * KG / 8;
  const int kchunks = KG / 8;
  for (size_t idx = (size_t)blockIdx.x * 256 + threadIdx.x; idx < total8;
       idx += (size_t)gridDim.x * 256) {
    int n = (int)(idx / kchunks);   // n = j*4 + g (gate-interleaved columns)
    int k = (int)(idx % kchunks) * 8;
    int j = n >> 2, g = n & 3;
    const float* src = (k < I_) ? (Wx + ((size_t)g * H_ + j) * I_ + k)
                                : (Wh + ((size_t)g * H_ + j) * H_ + (k - I_));
    *(v8bf*)(Wg + idx * 8) = cvt8(src);
  }
}

__global__ __launch_bounds__(256) void pack_wout_kernel(
    const float* __restrict__ Wout, bf16_t* __restrict__ Woutb) {
  const size_t total8 = (size_t)O_ * H_ / 8;
  for (size_t idx = (size_t)blockIdx.x * 256 + threadIdx.x; idx < total8;
       idx += (size_t)gridDim.x * 256) {
    *(v8bf*)(Woutb + idx * 8) = cvt8(Wout + idx * 8);
  }
}

// ------------------------- TDM tile load (2D) ------------------------------
// Loads a rows x 32-element bf16 tile (rows stride K elements in global) into
// LDS with 16B padding after each 64B row => 80B LDS row stride.
// D# groups built in SGPR tuples; issued with tensor_load_to_lds.

__device__ __forceinline__ void tdm_load_tile(const bf16_t* gtile,
                                              unsigned lds_addr, int K,
                                              int rows) {
  unsigned long long ga = (unsigned long long)(size_t)gtile;
  v4i g0;
  g0[0] = 1;                                  // count=1 (valid), user mode
  g0[1] = (int)lds_addr;                      // lds_addr[31:0]
  g0[2] = (int)(unsigned)(ga & 0xFFFFFFFFu);  // global_addr[31:0]
  g0[3] = (int)(((unsigned)(ga >> 32) & 0x01FFFFFFu) | 0x80000000u); // ga[56:32]|type=2
  v8i g1;
  g1[0] = (1 << 16)     // data_size = 2 bytes
        | (1 << 20)     // pad_enable
        | (3 << 22)     // pad_interval: 16 dwords (64B) before padding
        | (3 << 25);    // pad_amount:   4 dwords (16B) of padding
  g1[1] = (32 << 16);               // tensor_dim0 = 32 (tile == tensor, no OOB)
  g1[2] = (rows << 16);             // tensor_dim0[31:16]=0, tensor_dim1 = rows
  g1[3] = (32 << 16);               // tensor_dim1[31:16]=0, tile_dim0 = 32
  g1[4] = rows;                     // tile_dim1 = rows, tile_dim2 = 0
  g1[5] = K;                        // tensor_dim0_stride[31:0] = K elements
  g1[6] = 0;                        // stride[47:32]=0, dim1_stride[15:0]=0
  g1[7] = 0;
  v4i gz;
  gz[0] = 0; gz[1] = 0; gz[2] = 0; gz[3] = 0;
  asm volatile("tensor_load_to_lds %0, %1, %2, %3"
               :: "s"(g0), "s"(g1), "s"(gz), "s"(gz)
               : "memory");
}

// ------------------------------ GEMM kernel --------------------------------
// 256 threads = 8 wave32 waves. Block tile 256(M) x 128(N), K-step 32,
// double-buffered LDS tiles fed by the Tensor Data Mover.
// Wave grid 4(M) x 2(N); each wave: 4x4 subtiles of 16x16 -> 16 v8f accums.

struct Smem {
  union {
    struct {
      bf16_t A[2][256 * 40];   // 2 buffers, 256 rows x 32 K, 80B padded rows
      bf16_t B[2][128 * 40];
    } st;                      // 61440 B
    float epi[64 * 132];       // epilogue shuffle buffer (33792 B)
  } u;
};

union F16x { v16bf v; v8bf h[2]; };

template <int EPI>  // EPI=1: fused LSTM gates epilogue; EPI=0: +bout epilogue
__global__ __launch_bounds__(256) void gemm_wmma(
    const bf16_t* __restrict__ Ag, const bf16_t* __restrict__ Bg,
    int K, int Ncols,
    const float* __restrict__ bx, const float* __restrict__ cin,
    float* __restrict__ hnew_f, bf16_t* __restrict__ hnew_b,
    float* __restrict__ outp, const float* __restrict__ bout) {
  __shared__ Smem s;
  const int tid  = threadIdx.x;
  const int lane = tid & 31;
  const int wave = tid >> 5;
  const int wm   = wave >> 1;       // 0..3  (M direction, 64 rows each)
  const int wn   = wave & 1;        // 0..1  (N direction, 64 cols each)
  const int l16  = lane & 15;
  const int kh   = lane >> 4;       // which K-half this lane owns

  const int m0 = blockIdx.x * 256;
  const int n0 = blockIdx.y * 128;

  v8f acc[4][4];
#pragma unroll
  for (int i = 0; i < 4; ++i)
#pragma unroll
    for (int j = 0; j < 4; ++j)
      acc[i][j] = v8f{0.f, 0.f, 0.f, 0.f, 0.f, 0.f, 0.f, 0.f};

  const unsigned ldsA[2] = {(unsigned)(size_t)&s.u.st.A[0][0],
                            (unsigned)(size_t)&s.u.st.A[1][0]};
  const unsigned ldsB[2] = {(unsigned)(size_t)&s.u.st.B[0][0],
                            (unsigned)(size_t)&s.u.st.B[1][0]};
  if (wave == 0) {  // TDM prologue: tiles 0 and 1 in flight
    tdm_load_tile(Ag + (size_t)m0 * K, ldsA[0], K, 256);
    tdm_load_tile(Bg + (size_t)n0 * K, ldsB[0], K, 128);
    if (K > 32) {
      tdm_load_tile(Ag + (size_t)m0 * K + 32, ldsA[1], K, 256);
      tdm_load_tile(Bg + (size_t)n0 * K + 32, ldsB[1], K, 128);
    }
  }

  for (int k0 = 0, it = 0; k0 < K; k0 += 32, ++it) {
    const int buf = it & 1;
    if (wave == 0) {  // wait for tile `it` (next tile's 2 ops may stay in flight)
      if (k0 + 32 < K) __builtin_amdgcn_s_wait_tensorcnt(2);
      else             __builtin_amdgcn_s_wait_tensorcnt(0);
    }
    if (wave == 1 && k0 + 64 < K)  // warm L2 for tile it+2
      __builtin_prefetch(Ag + (size_t)(m0 + lane * 8) * K + k0 + 64, 0, 3);
    __syncthreads();   // tile `buf` visible to all waves

    // ---- fragments (ISA 16-bit A 16x32 / B 32x16 lane layouts) ----
    // A lane(m=l16, kh): K in [8*kh, 8*kh+8) and [16+8*kh, 16+8*kh+8)
    // B lane(n=l16, kh): K in [8*kh, 8*kh+8) and [16+8*kh, 16+8*kh+8)
    F16x afr[4], bfr[4];
#pragma unroll
    for (int i = 0; i < 4; ++i) {
      int m = wm * 64 + i * 16 + l16;
      const bf16_t* p = &s.u.st.A[buf][m * 40 + kh * 8];
      afr[i].h[0] = *(const v8bf*)p;
      afr[i].h[1] = *(const v8bf*)(p + 16);
    }
#pragma unroll
    for (int j = 0; j < 4; ++j) {
      int n = wn * 64 + j * 16 + l16;
      const bf16_t* p = &s.u.st.B[buf][n * 40 + kh * 8];
      bfr[j].h[0] = *(const v8bf*)p;
      bfr[j].h[1] = *(const v8bf*)(p + 16);
    }
#pragma unroll
    for (int i = 0; i < 4; ++i)
#pragma unroll
      for (int j = 0; j < 4; ++j)
        acc[i][j] = __builtin_amdgcn_wmma_f32_16x16x32_bf16(
            false, afr[i].v, false, bfr[j].v, (short)0, acc[i][j], false, false);
    __syncthreads();   // all reads of `buf` done before TDM overwrites it

    if (wave == 0 && k0 + 64 < K) {  // issue tile it+2 into freed buffer
      tdm_load_tile(Ag + (size_t)m0 * K + (k0 + 64), ldsA[buf], K, 256);
      tdm_load_tile(Bg + (size_t)n0 * K + (k0 + 64), ldsB[buf], K, 128);
    }
  }

  if (EPI == 0) {
    // ---- plain epilogue: out = acc + bout ----
#pragma unroll
    for (int i = 0; i < 4; ++i)
#pragma unroll
      for (int j = 0; j < 4; ++j)
#pragma unroll
        for (int v = 0; v < 8; ++v) {
          int m = m0 + wm * 64 + i * 16 + v + 8 * kh;
          int n = n0 + wn * 64 + j * 16 + l16;
          outp[(size_t)m * Ncols + n] = acc[i][j][v] + bout[n];
        }
  } else {
    // ---- fused LSTM epilogue: regroup 4 gates/unit via LDS, 4 passes of
    // 64 rows; pass p is stored by the two waves with wm == p ----
    for (int p = 0; p < 4; ++p) {
      if (wm == p) {
#pragma unroll
        for (int i = 0; i < 4; ++i)
#pragma unroll
          for (int j = 0; j < 4; ++j)
#pragma unroll
            for (int v = 0; v < 8; ++v) {
              int r = i * 16 + v + 8 * kh;             // 0..63
              int n = wn * 64 + j * 16 + l16;          // 0..127
              s.u.epi[r * 132 + n] = acc[i][j][v];
            }
      }
      __syncthreads();
      for (int itx = tid; itx < 64 * 32; itx += 256) {
        int j = itx & 31;           // unit within tile (32 units per tile)
        int r = itx >> 5;           // row within pass
        int b  = m0 + p * 64 + r;   // batch index
        int jg = (n0 >> 2) + j;     // global unit index
        const float* gp = &s.u.epi[r * 132 + j * 4];
        float gi = fsig(gp[0] + bx[0 * H_ + jg]);
        float go = fsig(gp[1] + bx[1 * H_ + jg]);
        float gf = fsig(gp[2] + bx[2 * H_ + jg]);
        float gz = ftanh_(gp[3] + bx[3 * H_ + jg]);
        float cn = gi * gz + gf * cin[(size_t)b * H_ + jg];
        float hn = go * ftanh_(cn);
        hnew_f[(size_t)b * H_ + jg] = hn;
        hnew_b[(size_t)b * H_ + jg] = f2bf(hn);
      }
      __syncthreads();
    }
  }
}

// ------------------------------- launch ------------------------------------

extern "C" void kernel_launch(void* const* d_in, const int* in_sizes, int n_in,
                              void* d_out, int out_size, void* d_ws, size_t ws_size,
                              hipStream_t stream) {
  const float* inp  = (const float*)d_in[0];
  const float* h    = (const float*)d_in[1];
  const float* c    = (const float*)d_in[2];
  const float* Wx   = (const float*)d_in[3];
  const float* bx   = (const float*)d_in[4];
  const float* Wh   = (const float*)d_in[5];
  const float* Wout = (const float*)d_in[6];
  const float* bout = (const float*)d_in[7];
  float* out = (float*)d_out;                 // [B*O] out, then [B*H] h_new

  char* ws = (char*)d_ws;
  const size_t off_Apack = 0;                                  // 48 MB
  const size_t off_Wg    = off_Apack + (size_t)B_ * KG * 2;    // 12 MB
  const size_t off_Woutb = off_Wg    + (size_t)NG * KG * 2;    //  1 MB
  const size_t off_Hnewb = off_Woutb + (size_t)O_ * H_ * 2;    // 32 MB
  bf16_t* Apack = (bf16_t*)(ws + off_Apack);
  bf16_t* Wg    = (bf16_t*)(ws + off_Wg);
  bf16_t* Woutb = (bf16_t*)(ws + off_Woutb);
  bf16_t* Hnewb = (bf16_t*)(ws + off_Hnewb);

  pack_a_kernel<<<2048, 256, 0, stream>>>(inp, h, Apack);
  pack_w_kernel<<<1024, 256, 0, stream>>>(Wx, Wh, Wg);
  pack_wout_kernel<<<64, 256, 0, stream>>>(Wout, Woutb);

  // gates GEMM + fused LSTM pointwise update
  dim3 g1(B_ / 256, NG / 128);   // 64 x 32
  gemm_wmma<1><<<g1, 256, 0, stream>>>(Apack, Wg, KG, NG,
                                       bx, c,
                                       out + (size_t)B_ * O_, Hnewb,
                                       nullptr, nullptr);

  // output GEMM: out = h_new @ Wout^T + bout
  dim3 g2(B_ / 256, O_ / 128);   // 64 x 4
  gemm_wmma<0><<<g2, 256, 0, stream>>>(Hnewb, Woutb, H_, O_,
                                       nullptr, nullptr,
                                       nullptr, nullptr,
                                       out, bout);
}
